// NATLayer_83854941487363
// MI455X (gfx1250) — matrix operand order repaired and checked by reading, not compiled, not run on hardware
//
#include <hip/hip_runtime.h>

// ---- problem constants (match reference) ----
#define B_  2
#define H_  56
#define W_  56
#define C_  256
#define NH_ 8
#define DH_ 32
#define K_  7
#define P_  3
#define HP_ 62
#define WP_ 62

typedef _Float16 h16;
typedef _Float16 v16h __attribute__((ext_vector_type(16)));
typedef _Float16 v8h  __attribute__((ext_vector_type(8)));
typedef float    v8f  __attribute__((ext_vector_type(8)));

// ------------------------------------------------------------------
// fp32 -> f16 weight conversion
// ------------------------------------------------------------------
__global__ void cvt_f32_to_f16(const float* __restrict__ src, h16* __restrict__ dst, int n) {
    int i = blockIdx.x * 256 + threadIdx.x;
    if (i < n) dst[i] = (h16)src[i];
}

// ------------------------------------------------------------------
// LayerNorm over C=256, one pixel per block (256 threads).
// wrap_pad=1: output grid is the (B,62,62) wrap-padded grid, source pixel
// computed with wrap indexing (fuses jnp.pad(mode='wrap') with LN1).
// ------------------------------------------------------------------
__global__ __launch_bounds__(256)
void ln_kernel(const float* __restrict__ x, const float* __restrict__ gamma,
               const float* __restrict__ beta, h16* __restrict__ out, int wrap_pad) {
    __shared__ float red[256];
    int p = blockIdx.x;
    int c = threadIdx.x;
    int src;
    if (wrap_pad) {
        int b  = p / (HP_ * WP_);
        int r  = p % (HP_ * WP_);
        int ip = r / WP_, jp = r % WP_;
        int i  = (ip - P_ + H_) % H_;
        int j  = (jp - P_ + W_) % W_;
        src = ((b * H_ + i) * W_ + j) * C_ + c;
    } else {
        src = p * C_ + c;
    }
    float v = x[src];
    red[c] = v; __syncthreads();
    for (int s = 128; s > 0; s >>= 1) { if (c < s) red[c] += red[c + s]; __syncthreads(); }
    float mu = red[0] * (1.0f / C_);
    __syncthreads();
    float d = v - mu;
    red[c] = d * d; __syncthreads();
    for (int s = 128; s > 0; s >>= 1) { if (c < s) red[c] += red[c + s]; __syncthreads(); }
    float var = red[0] * (1.0f / C_);
    float xn  = d * rsqrtf(var + 1e-5f) * gamma[c] + beta[c];
    out[p * C_ + c] = (h16)xn;
}

// ------------------------------------------------------------------
// GEMM epilogue helper.
//   EPI 0: store f16          (QKV)
//   EPI 1: leaky-relu, f16    (FC1)
//   EPI 2: f32 = res + v      (proj -> residual buf, FC2 -> d_out)
// GUARD=false: whole 128-row block in range -> no per-element compare.
// ------------------------------------------------------------------
template<int EPI, bool GUARD>
__device__ __forceinline__
void gemm_epi(const v8f (&acc)[2][4], const float* __restrict__ bias,
              const float* __restrict__ res, h16* __restrict__ Oh,
              float* __restrict__ Of, int M, int N,
              int bm, int bn, int wm, int wn, int g, int l16) {
    #pragma unroll
    for (int i = 0; i < 2; i++) {
        const int row0 = bm + wm * 32 + i * 16 + 8 * g;   // first D row this half-lane owns
        #pragma unroll
        for (int j = 0; j < 4; j++) {
            const int gn = bn + wn * 64 + j * 16 + l16;
            const float bv = bias[gn];
            int off = row0 * N + gn;                       // 32-bit offsets throughout
            #pragma unroll
            for (int e = 0; e < 8; e++, off += N) {
                if (!GUARD || (row0 + e) < M) {
                    float v = acc[i][j][e] + bv;
                    if (EPI == 0)      Oh[off] = (h16)v;
                    else if (EPI == 1) { v = (v >= 0.f) ? v : 0.01f * v; Oh[off] = (h16)v; }
                    else               Of[off] = res[off] + v;
                }
            }
        }
    }
}

// ------------------------------------------------------------------
// WMMA GEMM: C[M,N] = A[M,K] @ W[N,K]^T + bias.
// Block tile 128x128, 8 waves, wave tile 32(M) x 64(N) = 2x4 WMMA subtiles.
// Double-buffered K loop: next 32-wide slab prefetched to registers while
// the current slab feeds 8 v_wmma_f32_16x16x32_f16.
// ------------------------------------------------------------------
template<int EPI>
__global__ __launch_bounds__(256)
void gemm_wmma(const h16* __restrict__ A, const h16* __restrict__ Wt,
               const float* __restrict__ bias, const float* __restrict__ res,
               h16* __restrict__ Oh, float* __restrict__ Of,
               int M, int N, int K) {
    constexpr int LDA = 40;             // halfs per LDS row (80B: conflict-free b128)
    __shared__ h16 lA[128 * LDA];
    __shared__ h16 lB[128 * LDA];

    const int tid  = threadIdx.x;
    const int lane = tid & 31;
    const int wave = tid >> 5;
    const int wm   = wave & 3;          // M-subtile group (0..3)
    const int wn   = wave >> 2;         // N-subtile group (0..1)
    const int g    = lane >> 4;         // lane half-group
    const int l16  = lane & 15;

    const int bm = blockIdx.x * 128;
    const int bn = blockIdx.y * 128;

    v8f acc[2][4];
    #pragma unroll
    for (int i = 0; i < 2; i++)
        #pragma unroll
        for (int j = 0; j < 4; j++) acc[i][j] = v8f{};

    const int lr = tid >> 1;            // 0..127: tile row this thread stages
    const int lc = (tid & 1) * 16;      // half-offset within the 32-wide K slab
    int ga_row = bm + lr; if (ga_row >= M) ga_row = 0;   // clamped read, store guarded
    const int gb_row = bn + lr;                          // N always multiple of 128

    const uint4* gA = (const uint4*)(A  + (long)ga_row * K);
    const uint4* gB = (const uint4*)(Wt + (long)gb_row * K);

    // prologue: prefetch K-slab 0
    uint4 ra0 = gA[lc >> 3], ra1 = gA[(lc + 8) >> 3];
    uint4 rb0 = gB[lc >> 3], rb1 = gB[(lc + 8) >> 3];

    for (int kk = 0; kk < K; kk += 32) {
        *(uint4*)&lA[lr * LDA + lc]     = ra0;
        *(uint4*)&lA[lr * LDA + lc + 8] = ra1;
        *(uint4*)&lB[lr * LDA + lc]     = rb0;
        *(uint4*)&lB[lr * LDA + lc + 8] = rb1;
        __syncthreads();

        const int kn = kk + 32;
        if (kn < K) {                   // uniform: prefetch next slab during WMMAs
            ra0 = gA[(kn + lc) >> 3];  ra1 = gA[(kn + lc + 8) >> 3];
            rb0 = gB[(kn + lc) >> 3];  rb1 = gB[(kn + lc + 8) >> 3];
        }

        // A fragment (16x32 f16, ISA 7.12.2): lane m=l16, K chunks [8g,8g+8), [16+8g,24+8g)
        v16h af[2], bf[4];
        #pragma unroll
        for (int i = 0; i < 2; i++) {
            const h16* ar = &lA[(wm * 32 + i * 16 + l16) * LDA];
            v8h lo = *(const v8h*)(ar + 8 * g);
            v8h hi = *(const v8h*)(ar + 16 + 8 * g);
            af[i] = __builtin_shufflevector(lo, hi, 0,1,2,3,4,5,6,7,8,9,10,11,12,13,14,15);
        }
        // B fragment (32x16): lane n=l16 holds K chunks [8g,8g+8), [16+8g,24+8g)
        #pragma unroll
        for (int j = 0; j < 4; j++) {
            const h16* br = &lB[(wn * 64 + j * 16 + l16) * LDA];
            v8h lo = *(const v8h*)(br + 8 * g);
            v8h hi = *(const v8h*)(br + 16 + 8 * g);
            bf[j] = __builtin_shufflevector(lo, hi, 0,1,2,3,4,5,6,7,8,9,10,11,12,13,14,15);
        }
        #pragma unroll
        for (int i = 0; i < 2; i++)
            #pragma unroll
            for (int j = 0; j < 4; j++)
                acc[i][j] = __builtin_amdgcn_wmma_f32_16x16x32_f16(
                    false, af[i], false, bf[j], (short)0, acc[i][j], false, false);
        __syncthreads();
    }

    if (bm + 128 <= M)   // uniform branch: fast path has zero per-element guards
        gemm_epi<EPI, false>(acc, bias, res, Oh, Of, M, N, bm, bn, wm, wn, g, l16);
    else
        gemm_epi<EPI, true >(acc, bias, res, Oh, Of, M, N, bm, bn, wm, wn, g, l16);
}

// ------------------------------------------------------------------
// Neighborhood attention, flash-style online softmax.
// One thread per (pixel, head): 49 neighbors x DH=32 f16 loads, f32 math.
// ------------------------------------------------------------------
__global__ __launch_bounds__(256)
void natten_kernel(const h16* __restrict__ qkv, const float* __restrict__ rpb,
                   h16* __restrict__ out) {
    int idx = blockIdx.x * 256 + threadIdx.x;
    if (idx >= B_ * H_ * W_ * NH_) return;
    const int h = idx & (NH_ - 1);
    const int p = idx >> 3;
    const int j = p % W_;
    const int i = (p / W_) % H_;
    const int b = p / (H_ * W_);

    const h16* qp = qkv + ((b * HP_ + i + P_) * WP_ + (j + P_)) * (3 * C_) + h * DH_;
    float q[DH_];
    #pragma unroll
    for (int d = 0; d < DH_; d++) q[d] = (float)qp[d] * 0.17677669529663687f; // DH^-0.5

    float m = -1e30f, l = 0.f;
    float o[DH_];
    #pragma unroll
    for (int d = 0; d < DH_; d++) o[d] = 0.f;

    for (int di = 0; di < K_; di++) {
        for (int dj = 0; dj < K_; dj++) {
            const h16* kp = qkv + ((b * HP_ + i + di) * WP_ + (j + dj)) * (3 * C_)
                                + C_ + h * DH_;
            float s = rpb[h * 169 + (P_ + di) * 13 + (P_ + dj)];
            #pragma unroll
            for (int d = 0; d < DH_; d++) s += q[d] * (float)kp[d];
            float mn = fmaxf(m, s);
            float sc = __expf(m - mn);
            float w  = __expf(s - mn);
            l = l * sc + w;
            const h16* vp = kp + C_;
            #pragma unroll
            for (int d = 0; d < DH_; d++) o[d] = o[d] * sc + w * (float)vp[d];
            m = mn;
        }
    }
    const float inv = 1.f / l;
    h16* op = out + p * C_ + h * DH_;
    #pragma unroll
    for (int d = 0; d < DH_; d++) op[d] = (h16)(o[d] * inv);
}

// ------------------------------------------------------------------
// launch
// ------------------------------------------------------------------
extern "C" void kernel_launch(void* const* d_in, const int* in_sizes, int n_in,
                              void* d_out, int out_size, void* d_ws, size_t ws_size,
                              hipStream_t stream) {
    const float* x      = (const float*)d_in[0];
    const float* gamma1 = (const float*)d_in[1];
    const float* beta1  = (const float*)d_in[2];
    const float* qkv_w  = (const float*)d_in[3];
    const float* qkv_b  = (const float*)d_in[4];
    const float* rpb    = (const float*)d_in[5];
    const float* proj_w = (const float*)d_in[6];
    const float* proj_b = (const float*)d_in[7];
    const float* gamma2 = (const float*)d_in[8];
    const float* beta2  = (const float*)d_in[9];
    const float* fc1_w  = (const float*)d_in[10];
    const float* fc1_b  = (const float*)d_in[11];
    const float* fc2_w  = (const float*)d_in[12];
    const float* fc2_b  = (const float*)d_in[13];
    float* out = (float*)d_out;

    const long NPAD  = (long)B_ * HP_ * WP_;   // 7688 padded pixels
    const long NPIX  = (long)B_ * H_  * W_;    // 6272 pixels

    // workspace layout (bytes)
    char* ws = (char*)d_ws;
    size_t off = 0;
    auto alloc = [&](size_t bytes) { char* p = ws + off; off += (bytes + 255) & ~(size_t)255; return p; };
    h16*   xnpad  = (h16*)  alloc(NPAD * C_ * 2);            // LN1 of wrap-padded grid, f16
    h16*   wq     = (h16*)  alloc((size_t)768 * 256 * 2);    // qkv_w f16
    h16*   wproj  = (h16*)  alloc((size_t)256 * 256 * 2);
    h16*   wfc1   = (h16*)  alloc((size_t)1024 * 256 * 2);
    h16*   wfc2   = (h16*)  alloc((size_t)256 * 1024 * 2);
    h16*   qkvbuf = (h16*)  alloc(NPAD * 768 * 2);           // QKV on padded grid, f16
    h16*   attn_o = (h16*)  alloc(NPIX * C_ * 2);            // attention output, f16
    float* resbuf = (float*)alloc(NPIX * C_ * 4);            // x + attn proj (residual), f32
    h16*   h2n    = (h16*)  alloc(NPIX * C_ * 2);            // LN2 output, f16
    h16*   fc1out = (h16*)  alloc(NPIX * 1024 * 2);          // FC1 output, f16

    // 1) weights -> f16
    cvt_f32_to_f16<<<(768 * 256 + 255) / 256, 256, 0, stream>>>(qkv_w,  wq,    768 * 256);
    cvt_f32_to_f16<<<(256 * 256 + 255) / 256, 256, 0, stream>>>(proj_w, wproj, 256 * 256);
    cvt_f32_to_f16<<<(1024 * 256 + 255) / 256, 256, 0, stream>>>(fc1_w, wfc1, 1024 * 256);
    cvt_f32_to_f16<<<(256 * 1024 + 255) / 256, 256, 0, stream>>>(fc2_w, wfc2, 256 * 1024);

    // 2) LN1 fused with wrap padding -> (B,62,62,256) f16
    ln_kernel<<<(int)NPAD, 256, 0, stream>>>(x, gamma1, beta1, xnpad, 1);

    // 3) QKV GEMM: 7688 x 768 x 256
    gemm_wmma<0><<<dim3(61, 6), 256, 0, stream>>>(xnpad, wq, qkv_b, nullptr,
                                                  qkvbuf, nullptr, (int)NPAD, 768, 256);

    // 4) neighborhood attention
    natten_kernel<<<(B_ * H_ * W_ * NH_ + 255) / 256, 256, 0, stream>>>(qkvbuf, rpb, attn_o);

    // 5) proj GEMM + residual: res = x + attn @ proj_w^T + b  (6272 x 256 x 256)
    gemm_wmma<2><<<dim3(49, 2), 256, 0, stream>>>(attn_o, wproj, proj_b, x,
                                                  nullptr, resbuf, (int)NPIX, 256, 256);

    // 6) LN2
    ln_kernel<<<(int)NPIX, 256, 0, stream>>>(resbuf, gamma2, beta2, h2n, 0);

    // 7) FC1 + leaky relu (6272 x 1024 x 256)
    gemm_wmma<1><<<dim3(49, 8), 256, 0, stream>>>(h2n, wfc1, fc1_b, nullptr,
                                                  fc1out, nullptr, (int)NPIX, 1024, 256);

    // 8) FC2 + residual -> d_out (6272 x 256 x 1024)
    gemm_wmma<2><<<dim3(49, 2), 256, 0, stream>>>(fc1out, wfc2, fc2_b, resbuf,
                                                  nullptr, out, (int)NPIX, 256, 1024);
}
